// FBABRF_Encoder_35003983462534
// MI455X (gfx1250) — matrix operand order repaired and checked by reading, not compiled, not run on hardware
//
#include <hip/hip_runtime.h>
#include <hip/hip_bf16.h>

// ---------------- problem constants ----------------
#define Bsz 256
#define Lsz 256
#define Dd  128
#define Hh  2
#define Ee  64
#define NLn 2
#define TOPKn 5
#define BLn (Bsz*Lsz)          // 65536
#define BLDn ((size_t)BLn*Dd)  // 8388608

typedef __attribute__((ext_vector_type(16))) __bf16 v16bf;
typedef __attribute__((ext_vector_type(8)))  float v8f;
typedef __attribute__((ext_vector_type(8)))  unsigned int uint8v;
typedef int v4i32 __attribute__((vector_size(16)));

#define ASL __attribute__((address_space(3)))
#define ASG __attribute__((address_space(1)))

#if defined(__has_builtin)
#if __has_builtin(__builtin_amdgcn_global_load_async_to_lds_b128) && \
    __has_builtin(__builtin_amdgcn_s_wait_asynccnt)
#define USE_ASYNC_LDS 1
#else
#define USE_ASYNC_LDS 0
#endif
#else
#define USE_ASYNC_LDS 0
#endif

#if USE_ASYNC_LDS
#define ASYNC_CP(src, dst) __builtin_amdgcn_global_load_async_to_lds_b128( \
        (ASG v4i32*)(src), (ASL v4i32*)(dst), 0, 0)
#define ASYNC_WAIT(n) __builtin_amdgcn_s_wait_asynccnt(n)
#else
#define ASYNC_CP(src, dst) (*(uint4*)(dst) = *(const uint4*)(src))
#define ASYNC_WAIT(n) ((void)0)
#endif

__device__ __forceinline__ float bf2f(unsigned short u) {
    unsigned int x = ((unsigned int)u) << 16;
    return __builtin_bit_cast(float, x);
}
__device__ __forceinline__ unsigned short f2bf(float f) {
    unsigned int u = __builtin_bit_cast(unsigned int, f);
    unsigned int r = u + 0x7FFFu + ((u >> 16) & 1u);
    return (unsigned short)(r >> 16);
}

// ISA 16-bit A-operand per-lane layout: K = {8g..8g+7} U {16+8g..16+8g+7}
__device__ __forceinline__ v16bf fragA(const unsigned short* row, int g) {
    uint4 x0 = *(const uint4*)(row + 8 * g);
    uint4 x1 = *(const uint4*)(row + 16 + 8 * g);
    uint8v u = {x0.x, x0.y, x0.z, x0.w, x1.x, x1.y, x1.z, x1.w};
    return __builtin_bit_cast(v16bf, u);
}
// ISA 16-bit B-operand per-lane layout: K = {8g..8g+7} U {16+8g..16+8g+7} (from N-row of BT)
__device__ __forceinline__ v16bf fragB(const unsigned short* row, int g) {
    uint4 x0 = *(const uint4*)(row + 8 * g);
    uint4 x1 = *(const uint4*)(row + 16 + 8 * g);
    uint8v u = {x0.x, x0.y, x0.z, x0.w, x1.x, x1.y, x1.z, x1.w};
    return __builtin_bit_cast(v16bf, u);
}

// ---------------- generic batched WMMA GEMM (BT form) ----------------
// C[z] (MxN) = A[z] (MxK bf16 row-major) * BT[z]^T, BT is N x K row-major
// (K contiguous for BOTH operands -> pure async b128 staging, no LDS scatter).
// Block: 128 threads (4 waves), tile 64(M) x 64(N), K-step 32, double-buffered.
// Each wave: 32x32 output = 2 A-frags x 2 B-frags = 4 wmma per K-step.
// storeMode: 0 = f32 C[m][n], 1 = bf16 C[m][n], 2 = bf16 transposed C[n][m]
__global__ __launch_bounds__(128)
void wmma_gemm_bf16(const unsigned short* __restrict__ A,
                    const unsigned short* __restrict__ Bt,
                    void* __restrict__ C,
                    int M, int N, int K, int lda, int ldbt, int ldc,
                    long long sAb, long long sAh,
                    long long sBb, long long sBh,
                    long long sCb, long long sCh,
                    int Hdim, int storeMode)
{
    (void)M; (void)N;
    int z  = blockIdx.z;
    int bb = z / Hdim;
    int hh = z - bb * Hdim;
    const unsigned short* Ap  = A  + (size_t)bb * sAb + (size_t)hh * sAh;
    const unsigned short* Btp = Bt + (size_t)bb * sBb + (size_t)hh * sBh;
    size_t coff = (size_t)bb * sCb + (size_t)hh * sCh;

    __shared__ unsigned short As [2][64][40];   // 80B pitch
    __shared__ unsigned short Bts[2][64][40];

    int tid   = threadIdx.x;
    int lane  = tid & 31;
    int w     = tid >> 5;
    int mlane = lane & 15;
    int g     = lane >> 4;
    int wm    = w >> 1;                         // 2x2 waves over 64x64
    int wn    = w & 1;
    int rowBase = blockIdx.y * 64;
    int colBase = blockIdx.x * 64;

    int sr = tid >> 1;                          // staging row 0..63
    int sc = (tid & 1) * 16;                    // K sub-chunk 0 / 16
    const unsigned short* gArow = Ap  + (size_t)(rowBase + sr) * lda  + sc;
    const unsigned short* gBrow = Btp + (size_t)(colBase + sr) * ldbt + sc;

    v8f acc00 = {0.f,0.f,0.f,0.f,0.f,0.f,0.f,0.f};
    v8f acc01 = acc00, acc10 = acc00, acc11 = acc00;

    const int nT = K >> 5;
    auto issueTile = [&](int t, int s) {
        const unsigned short* ga = gArow + t * 32;
        const unsigned short* gb = gBrow + t * 32;
        ASYNC_CP(ga,     &As [s][sr][sc]);
        ASYNC_CP(ga + 8, &As [s][sr][sc + 8]);
        ASYNC_CP(gb,     &Bts[s][sr][sc]);
        ASYNC_CP(gb + 8, &Bts[s][sr][sc + 8]);
    };

    issueTile(0, 0);
    for (int t = 0; t < nT; ++t) {
        int s = t & 1;
        if (t + 1 < nT) { issueTile(t + 1, s ^ 1); ASYNC_WAIT(4); }
        else            { ASYNC_WAIT(0); }
        __syncthreads();

        v16bf a0 = fragA(&As [s][wm * 32 + mlane][0],      g);
        v16bf a1 = fragA(&As [s][wm * 32 + 16 + mlane][0], g);
        v16bf b0 = fragB(&Bts[s][wn * 32 + mlane][0],      g);
        v16bf b1 = fragB(&Bts[s][wn * 32 + 16 + mlane][0], g);

        __syncthreads();   // reads done before buffer s is refilled (t+2)

        acc00 = __builtin_amdgcn_wmma_f32_16x16x32_bf16(false, a0, false, b0, (short)0, acc00, false, false);
        acc01 = __builtin_amdgcn_wmma_f32_16x16x32_bf16(false, a0, false, b1, (short)0, acc01, false, false);
        acc10 = __builtin_amdgcn_wmma_f32_16x16x32_bf16(false, a1, false, b0, (short)0, acc10, false, false);
        acc11 = __builtin_amdgcn_wmma_f32_16x16x32_bf16(false, a1, false, b1, (short)0, acc11, false, false);
    }

    auto storeTile = [&](v8f ac, int mi, int nj) {
        int cm = rowBase + wm * 32 + mi * 16 + 8 * g;  // D layout: M=r+8g, N=lane%16
        int cn = colBase + wn * 32 + nj * 16 + mlane;
        if (storeMode == 0) {
            float* Cf = (float*)C + coff;
#pragma unroll
            for (int r = 0; r < 8; r++) Cf[(size_t)(cm + r) * ldc + cn] = ac[r];
        } else if (storeMode == 1) {
            unsigned short* Cb = (unsigned short*)C + coff;
#pragma unroll
            for (int r = 0; r < 8; r++) Cb[(size_t)(cm + r) * ldc + cn] = f2bf(ac[r]);
        } else {
            unsigned short* Cb = (unsigned short*)C + coff;
#pragma unroll
            for (int r = 0; r < 8; r++) Cb[(size_t)cn * ldc + (cm + r)] = f2bf(ac[r]);
        }
    };
    storeTile(acc00, 0, 0); storeTile(acc01, 0, 1);
    storeTile(acc10, 1, 0); storeTile(acc11, 1, 1);
}

// ---------------- glue kernels ----------------
__global__ void embed_kernel(const int* __restrict__ paths,
                             const float* __restrict__ ego,
                             const float* __restrict__ pos,
                             float* __restrict__ x,
                             unsigned short* __restrict__ xbf,
                             float* __restrict__ maskv)
{
    int bl = blockIdx.x;           // B*L
    int d  = threadIdx.x;          // 128
    int p  = paths[bl];
    int l  = bl & (Lsz - 1);
    float v = ego[(size_t)p * Dd + d] + pos[l * Dd + d];
    x  [(size_t)bl * Dd + d] = v;
    xbf[(size_t)bl * Dd + d] = f2bf(v);
    if (d == 0) maskv[bl] = (p < 100000) ? 0.0f : -10000.0f;
}

__global__ void f32_to_bf16_kernel(const float* __restrict__ s,
                                   unsigned short* __restrict__ d, size_t n)
{
    size_t i = (size_t)blockIdx.x * blockDim.x + threadIdx.x;
    if (i < n) d[i] = f2bf(s[i]);
}

// convert f32 weights (layers x K x N) to bf16 transposed (layers x N x K)
__global__ void w_transpose_kernel(const float* __restrict__ W,
                                   unsigned short* __restrict__ WT,
                                   int Kd, int Nd, size_t n)
{
    size_t i = (size_t)blockIdx.x * 256 + threadIdx.x;
    if (i >= n) return;
    size_t kn  = (size_t)Kd * Nd;
    size_t lay = i / kn;
    size_t rem = i - lay * kn;
    int kk = (int)(rem / Nd);
    int nn = (int)(rem - (size_t)kk * Nd);
    WT[lay * kn + (size_t)nn * Kd + kk] = f2bf(W[i]);
}

// band projection matrix: P[t,s] = (2/L) * sum_{f=lo}^{hi-1} a_f cos(2*pi*f*(t-s)/L)
// (P is symmetric, so it serves as both A and BT operand.)
__global__ void build_proj_kernel(unsigned short* __restrict__ P, int lo, int hi)
{
    int t = blockIdx.x, s = threadIdx.x;
    float ph = (float)(t - s) * (6.28318530718f / (float)Lsz);
    float acc = 0.0f;
    for (int f = lo; f < hi; f++) {
        float a = (f == 0 || f == Lsz / 2) ? 0.5f : 1.0f;
        acc += a * __cosf(ph * (float)f);
    }
    P[t * Lsz + s] = f2bf(acc * (2.0f / (float)Lsz));
}

__global__ void softmax_kernel(const float* __restrict__ S,
                               const float* __restrict__ maskv,
                               unsigned short* __restrict__ att)
{
    int row = blockIdx.x;               // B*H*L rows
    int c   = threadIdx.x;              // 256
    int b   = row >> 9;                 // / (H*L)
    __shared__ float red[256];
    float v = S[(size_t)row * Lsz + c] * 0.125f + maskv[b * Lsz + c];
    red[c] = v; __syncthreads();
    for (int o = 128; o > 0; o >>= 1) {
        if (c < o) red[c] = fmaxf(red[c], red[c + o]);
        __syncthreads();
    }
    float mx = red[0]; __syncthreads();
    float e = __expf(v - mx);
    red[c] = e; __syncthreads();
    for (int o = 128; o > 0; o >>= 1) {
        if (c < o) red[c] += red[c + o];
        __syncthreads();
    }
    float inv = 1.0f / red[0];
    att[(size_t)row * Lsz + c] = f2bf(e * inv);
}

// mean_value[b,tau] = (1/D) sum_{t,d} qt[b,(t+tau)%L,d] * kt[b,t,d]
__global__ void mean_corr_kernel(const unsigned short* __restrict__ qt,
                                 const unsigned short* __restrict__ kt,
                                 float* __restrict__ mv)
{
    int b = blockIdx.y;
    int tauBase = blockIdx.x * 4;
    int t = threadIdx.x;                 // 256
    __shared__ float red[256];
    const unsigned short* qb = qt + (size_t)b * Lsz * Dd;
    const unsigned short* kb = kt + (size_t)b * Lsz * Dd;
    for (int j = 0; j < 4; j++) {
        int tau = tauBase + j;
        int tq  = (t + tau) & (Lsz - 1);
        float s = 0.0f;
        for (int d = 0; d < Dd; d++)
            s += bf2f(qb[(size_t)tq * Dd + d]) * bf2f(kb[(size_t)t * Dd + d]);
        red[t] = s; __syncthreads();
        for (int o = 128; o > 0; o >>= 1) {
            if (t < o) red[t] += red[t + o];
            __syncthreads();
        }
        if (t == 0) mv[b * Lsz + tau] = red[0] * (1.0f / (float)Dd);
        __syncthreads();
    }
}

__global__ void colmean_kernel(const float* __restrict__ mv, float* __restrict__ colm)
{
    int tau = threadIdx.x;               // 256
    float s = 0.0f;
    for (int b = 0; b < Bsz; b++) s += mv[b * Lsz + tau];
    colm[tau] = s * (1.0f / (float)Bsz);
}

__global__ void topk_kernel(const float* __restrict__ colm, int* __restrict__ delays)
{
    if (threadIdx.x == 0 && blockIdx.x == 0) {
        float vals[Lsz];
        for (int i = 0; i < Lsz; i++) vals[i] = colm[i];
        for (int j = 0; j < TOPKn; j++) {
            int bi = 0; float bv = -1e30f;
            for (int i = 0; i < Lsz; i++)
                if (vals[i] > bv) { bv = vals[i]; bi = i; }
            delays[j] = bi;
            vals[bi] = -1e30f;
        }
    }
}

__global__ void wsm_kernel(const float* __restrict__ mv, const int* __restrict__ delays,
                           float* __restrict__ tmpc)
{
    int b = blockIdx.x;
    int j = threadIdx.x;
    __shared__ float w[TOPKn];
    if (j < TOPKn) w[j] = mv[b * Lsz + delays[j]];
    __syncthreads();
    if (j < TOPKn) {
        float mx = -1e30f;
        for (int i = 0; i < TOPKn; i++) mx = fmaxf(mx, w[i]);
        float s = 0.0f;
        for (int i = 0; i < TOPKn; i++) s += __expf(w[i] - mx);
        tmpc[b * TOPKn + j] = __expf(w[j] - mx) / s;
    }
}

// comb = 0.9*freq + 0.1*spatial, bf16; v is stored transposed (B, D, L)
__global__ void comb_kernel(const unsigned short* __restrict__ vT,
                            const float* __restrict__ spat,
                            const float* __restrict__ tmpc,
                            const int* __restrict__ delays,
                            unsigned short* __restrict__ comb)
{
    int bl = blockIdx.x;                 // B*L
    int d  = threadIdx.x;                // 128
    int b  = bl >> 8;
    int l  = bl & (Lsz - 1);
    float acc = 0.0f;
    const unsigned short* vrow = vT + ((size_t)b * Dd + d) * Lsz;
    for (int j = 0; j < TOPKn; j++) {
        int src = (l + delays[j]) & (Lsz - 1);
        acc += tmpc[b * TOPKn + j] * bf2f(vrow[src]);
    }
    float out = 0.9f * acc + 0.1f * spat[(size_t)bl * Dd + d];
    comb[(size_t)bl * Dd + d] = f2bf(out);
}

__global__ void residual_bf_kernel(const float* __restrict__ wpout,
                                   const float* __restrict__ x,
                                   unsigned short* __restrict__ abf, size_t n)
{
    size_t i = (size_t)blockIdx.x * 256 + threadIdx.x;
    if (i < n) abf[i] = f2bf(wpout[i] + x[i]);
}

__global__ void gelu_bf_kernel(const float* __restrict__ h,
                               unsigned short* __restrict__ o, size_t n)
{
    size_t i = (size_t)blockIdx.x * 256 + threadIdx.x;
    if (i < n) {
        float v = h[i];
        float g = v * 0.5f * (1.0f + erff(v * 0.70710678f));
        o[i] = f2bf(g);
    }
}

__global__ void gather_out_kernel(const float* __restrict__ x,
                                  const int* __restrict__ lengths,
                                  float* __restrict__ out)
{
    int b = blockIdx.x, d = threadIdx.x;
    int idx = lengths[b] - 1;
    if (idx < 0) idx = 0; if (idx > Lsz - 1) idx = Lsz - 1;
    out[b * Dd + d] = x[((size_t)b * Lsz + idx) * Dd + d];
}

// ---------------- host orchestration ----------------
extern "C" void kernel_launch(void* const* d_in, const int* in_sizes, int n_in,
                              void* d_out, int out_size, void* d_ws, size_t ws_size,
                              hipStream_t stream)
{
    (void)in_sizes; (void)n_in; (void)out_size; (void)ws_size;
    const int*   paths   = (const int*)d_in[0];
    const int*   lengths = (const int*)d_in[1];
    const float* ego     = (const float*)d_in[4];
    const float* pos     = (const float*)d_in[5];
    const float* Wq      = (const float*)d_in[6];
    const float* Wk      = (const float*)d_in[7];
    const float* Wv      = (const float*)d_in[8];
    const float* Wp      = (const float*)d_in[9];
    const float* F1      = (const float*)d_in[10];
    const float* F2      = (const float*)d_in[11];
    float* out = (float*)d_out;

    char* ws = (char*)d_ws;
    size_t off = 0;
    auto alloc = [&](size_t bytes) { size_t o = off; off += (bytes + 255) & ~(size_t)255; return o; };

    const size_t BLD = BLDn;
    size_t xF32   = alloc(BLD * 4);
    size_t xBF    = alloc(BLD * 2);
    size_t qBF    = alloc(BLD * 2);           // qT (B,D,L); reused as a_bf16
    size_t kBF    = alloc(BLD * 2);           // kT (B,D,L); reused as comb_bf16
    size_t vBF    = alloc(BLD * 2);           // vT (B,D,L)
    size_t qtBF   = alloc(BLD * 2);           // qt (B,L,D)
    size_t ktBF   = alloc(BLD * 2);           // kt (B,L,D)
    size_t vtBF   = alloc(BLD * 2);           // vtT (B,D,L)
    size_t Sf     = alloc((size_t)Bsz * Hh * Lsz * Lsz * 4);  // reused as h fp32
    size_t attB   = alloc((size_t)Bsz * Hh * Lsz * Lsz * 2);  // reused as h bf16
    size_t spatF  = alloc(BLD * 4);           // reused as Wp output
    size_t wAll   = alloc((size_t)(4 * NLn * Dd * Dd + 2 * NLn * Dd * 2 * Dd) * 2);
    size_t pBF    = alloc((size_t)Lsz * Lsz * 2);
    size_t meanV  = alloc((size_t)Bsz * Lsz * 4);
    size_t colM   = alloc((size_t)Lsz * 4);
    size_t delO   = alloc((size_t)TOPKn * 4);
    size_t tmpC   = alloc((size_t)Bsz * TOPKn * 4);
    size_t maskV  = alloc((size_t)Bsz * Lsz * 4);

    // all weights stored bf16 TRANSPOSED (N x K per layer)
    unsigned short* wqb = (unsigned short*)(ws + wAll);
    unsigned short* wkb = wqb + NLn * Dd * Dd;
    unsigned short* wvb = wkb + NLn * Dd * Dd;
    unsigned short* wpb = wvb + NLn * Dd * Dd;
    unsigned short* f1b = wpb + NLn * Dd * Dd;     // (2D x D) per layer
    unsigned short* f2b = f1b + NLn * Dd * 2 * Dd; // (D x 2D) per layer

    auto gemm = [&](const unsigned short* A, const unsigned short* Bt, void* C,
                    int M, int N, int K, int lda, int ldbt, int ldc,
                    long long sAb, long long sAh, long long sBb, long long sBh,
                    long long sCb, long long sCh, int Hdim, int batch, int mode) {
        dim3 g(N / 64, M / 64, batch);
        wmma_gemm_bf16<<<g, 128, 0, stream>>>(A, Bt, C, M, N, K, lda, ldbt, ldc,
                                              sAb, sAh, sBb, sBh, sCb, sCh, Hdim, mode);
    };

    // ---- one-time: weights -> bf16 transposed, embedding ----
    {
        size_t nw = (size_t)NLn * Dd * Dd;
        unsigned gw = (unsigned)((nw + 255) / 256);
        w_transpose_kernel<<<gw, 256, 0, stream>>>(Wq, wqb, Dd, Dd, nw);
        w_transpose_kernel<<<gw, 256, 0, stream>>>(Wk, wkb, Dd, Dd, nw);
        w_transpose_kernel<<<gw, 256, 0, stream>>>(Wv, wvb, Dd, Dd, nw);
        w_transpose_kernel<<<gw, 256, 0, stream>>>(Wp, wpb, Dd, Dd, nw);
        size_t nf = (size_t)NLn * Dd * 2 * Dd;
        unsigned gf = (unsigned)((nf + 255) / 256);
        w_transpose_kernel<<<gf, 256, 0, stream>>>(F1, f1b, Dd, 2 * Dd, nf);
        w_transpose_kernel<<<gf, 256, 0, stream>>>(F2, f2b, 2 * Dd, Dd, nf);
    }
    embed_kernel<<<BLn, Dd, 0, stream>>>(paths, ego, pos,
                                         (float*)(ws + xF32),
                                         (unsigned short*)(ws + xBF),
                                         (float*)(ws + maskV));

    const int LEFTS[NLn]  = {51, 0};
    const int RIGHTS[NLn] = {129, 78};
    const long long sLD = (long long)Lsz * Dd;   // == D*L

    for (int k = 0; k < NLn; k++) {
        build_proj_kernel<<<Lsz, Lsz, 0, stream>>>((unsigned short*)(ws + pBF),
                                                   LEFTS[k], RIGHTS[k]);
        // Q/K/V projections: per-b GEMM, output stored TRANSPOSED -> (B, D, L)
        gemm((unsigned short*)(ws + xBF), wqb + (size_t)k * Dd * Dd, ws + qBF,
             Lsz, Dd, Dd, Dd, Dd, Lsz, sLD, 0, 0, 0, sLD, 0, 1, Bsz, 2);
        gemm((unsigned short*)(ws + xBF), wkb + (size_t)k * Dd * Dd, ws + kBF,
             Lsz, Dd, Dd, Dd, Dd, Lsz, sLD, 0, 0, 0, sLD, 0, 1, Bsz, 2);
        gemm((unsigned short*)(ws + xBF), wvb + (size_t)k * Dd * Dd, ws + vBF,
             Lsz, Dd, Dd, Dd, Dd, Lsz, sLD, 0, 0, 0, sLD, 0, 1, Bsz, 2);
        // band filter: qt = P @ q, BT = qT (B,D,L); q/k normal out, v transposed out
        gemm((unsigned short*)(ws + pBF), (unsigned short*)(ws + qBF), ws + qtBF,
             Lsz, Dd, Lsz, Lsz, Lsz, Dd, 0, 0, sLD, 0, sLD, 0, 1, Bsz, 1);
        gemm((unsigned short*)(ws + pBF), (unsigned short*)(ws + kBF), ws + ktBF,
             Lsz, Dd, Lsz, Lsz, Lsz, Dd, 0, 0, sLD, 0, sLD, 0, 1, Bsz, 1);
        gemm((unsigned short*)(ws + pBF), (unsigned short*)(ws + vBF), ws + vtBF,
             Lsz, Dd, Lsz, Lsz, Lsz, Lsz, 0, 0, sLD, 0, sLD, 0, 1, Bsz, 2);
        // scores S = qt_h @ kt_h^T : BT is kt itself (N=L rows, K=E contiguous)
        gemm((unsigned short*)(ws + qtBF), (unsigned short*)(ws + ktBF), ws + Sf,
             Lsz, Lsz, Ee, Dd, Dd, Lsz,
             sLD, Ee, sLD, Ee,
             (long long)Hh * Lsz * Lsz, (long long)Lsz * Lsz, Hh, Bsz * Hh, 0);
        softmax_kernel<<<Bsz * Hh * Lsz, Lsz, 0, stream>>>(
            (float*)(ws + Sf), (float*)(ws + maskV), (unsigned short*)(ws + attB));
        // spatial = att @ vt_h : BT = vtT (N=E rows at offset h*E*L, K=L)
        gemm((unsigned short*)(ws + attB), (unsigned short*)(ws + vtBF), ws + spatF,
             Lsz, Ee, Lsz, Lsz, Lsz, Dd,
             (long long)Hh * Lsz * Lsz, (long long)Lsz * Lsz, sLD, (long long)Ee * Lsz,
             sLD, Ee, Hh, Bsz * Hh, 0);
        // autocorrelation delays
        mean_corr_kernel<<<dim3(Lsz / 4, Bsz), Lsz, 0, stream>>>(
            (unsigned short*)(ws + qtBF), (unsigned short*)(ws + ktBF), (float*)(ws + meanV));
        colmean_kernel<<<1, Lsz, 0, stream>>>((float*)(ws + meanV), (float*)(ws + colM));
        topk_kernel<<<1, 32, 0, stream>>>((float*)(ws + colM), (int*)(ws + delO));
        wsm_kernel<<<Bsz, 32, 0, stream>>>((float*)(ws + meanV), (int*)(ws + delO),
                                           (float*)(ws + tmpC));
        // comb = 0.9*freq + 0.1*spatial  (bf16, reuse kBF)
        comb_kernel<<<BLn, Dd, 0, stream>>>((unsigned short*)(ws + vBF),
                                            (float*)(ws + spatF), (float*)(ws + tmpC),
                                            (int*)(ws + delO), (unsigned short*)(ws + kBF));
        // Wp projection (fp32 into spatF region) + residual
        gemm((unsigned short*)(ws + kBF), wpb + (size_t)k * Dd * Dd, ws + spatF,
             BLn, Dd, Dd, Dd, Dd, Dd, 0, 0, 0, 0, 0, 0, 1, 1, 0);
        residual_bf_kernel<<<(unsigned)(BLD / 256), 256, 0, stream>>>(
            (float*)(ws + spatF), (float*)(ws + xF32), (unsigned short*)(ws + qBF), BLD);
        // FFN: h = a @ F1 (fp32 into S region), gelu -> bf16 (att region), x = h @ F2
        gemm((unsigned short*)(ws + qBF), f1b + (size_t)k * Dd * 2 * Dd, ws + Sf,
             BLn, 2 * Dd, Dd, Dd, Dd, 2 * Dd, 0, 0, 0, 0, 0, 0, 1, 1, 0);
        gelu_bf_kernel<<<(unsigned)((BLD * 2) / 256), 256, 0, stream>>>(
            (float*)(ws + Sf), (unsigned short*)(ws + attB), BLD * 2);
        gemm((unsigned short*)(ws + attB), f2b + (size_t)k * 2 * Dd * Dd, ws + xF32,
             BLn, Dd, 2 * Dd, 2 * Dd, 2 * Dd, Dd, 0, 0, 0, 0, 0, 0, 1, 1, 0);
        f32_to_bf16_kernel<<<(unsigned)(BLD / 256), 256, 0, stream>>>(
            (float*)(ws + xF32), (unsigned short*)(ws + xBF), BLD);
    }

    gather_out_kernel<<<Bsz, Dd, 0, stream>>>((float*)(ws + xF32), lengths, out);
}
